// PointerNet_33543694581840
// MI455X (gfx1250) — matrix operand order repaired and checked by reading, not compiled, not run on hardware
//
#include <hip/hip_runtime.h>
#include <stdint.h>

// Problem constants (match reference)
#define BB 128
#define SS 512
#define DD 256
#define HH 256
#define G4H 1024  // 4*H

typedef __attribute__((ext_vector_type(16))) __bf16 v16bf;
typedef __attribute__((ext_vector_type(8)))  float  v8f;

union Frag16 { v16bf v; uint32_t u[8]; };
union FB { float f; uint32_t u; };

__device__ __forceinline__ float bf2f(uint16_t x) {
    FB t; t.u = ((uint32_t)x) << 16; return t.f;
}
__device__ __forceinline__ uint16_t f2bf(float f) {
    FB t; t.f = f;
    uint32_t r = (t.u + 0x7FFFu + ((t.u >> 16) & 1u)) >> 16;  // RNE
    return (uint16_t)r;
}
__device__ __forceinline__ float sigmoidf_(float x) {
    return 1.0f / (1.0f + __expf(-x));
}

// ---- gfx1250 async global->LDS staging (feature-detected) --------------------
#ifndef __has_builtin
#define __has_builtin(x) 0
#endif
#if __has_builtin(__builtin_amdgcn_global_load_async_to_lds_b128)
#define HAVE_ASYNC_LDS 1
#else
#define HAVE_ASYNC_LDS 0
#endif

// Builtin expects pointers to 16B int vectors:
//   arg0: v4i addrspace(1)* (global), arg1: v4i addrspace(3)* (LDS)
typedef int v4i_t __attribute__((vector_size(16)));
typedef __attribute__((address_space(1))) v4i_t as1_v4i;
typedef __attribute__((address_space(3))) v4i_t as3_v4i;

struct U128 { uint32_t w[4]; };

__device__ __forceinline__ void stage16(const void* gsrc, void* ldst) {
#if HAVE_ASYNC_LDS
    as1_v4i* g = (as1_v4i*)(uintptr_t)gsrc;                 // global address
    as3_v4i* l = (as3_v4i*)(uint32_t)(uintptr_t)ldst;       // low 32b = LDS offset
    __builtin_amdgcn_global_load_async_to_lds_b128(g, l, 0, 0);
#else
    *(U128*)ldst = *(const U128*)gsrc;
#endif
}
__device__ __forceinline__ void wait_async0() {
#if HAVE_ASYNC_LDS
#if __has_builtin(__builtin_amdgcn_s_wait_asynccnt)
    __builtin_amdgcn_s_wait_asynccnt(0);
#else
    asm volatile("s_wait_asynccnt 0" ::: "memory");
#endif
#endif
}

// ---------------------------------------------------------------------------
// bf16 WMMA GEMM:  C[M,N] = A[M,K] @ Bw[N,K]^T (+ bias[N]) (+ CinitBf)
// Block = 8 waves; wave owns a 16x64 tile (A-frag reused across 4 WMMAs).
// B tile (64 cols x 32 k) staged through double-buffered LDS via async loads,
// shared by all 8 waves; A frags software-pipelined in registers.
// Requires exact tiling: M % 128 == 0 (grid covers), N % 64 == 0, K % 32 == 0.
// ---------------------------------------------------------------------------
#define BPAD 80  // bytes per B-tile row in LDS (64 data + 16 pad, 16B aligned)

__global__ __launch_bounds__(256) void gemm_bf16_wmma(
    const uint16_t* __restrict__ A, int lda,
    const uint16_t* __restrict__ Bw, int ldb,
    const float* __restrict__ bias,
    const uint16_t* __restrict__ CinitBf, long long ldcin,
    float* __restrict__ Cf, uint16_t* __restrict__ Cbf, long long ldc,
    int M, int N, int K)
{
    __shared__ __align__(16) uint8_t Btile[2][64 * BPAD];  // 2 x 5120 B

    const int tid  = threadIdx.x;
    const int lane = tid & 31;
    const int wave = tid >> 5;
    const int row0 = blockIdx.y * 128 + wave * 16;
    const int col0 = blockIdx.x * 64;
    const int half = lane >> 4;   // 0 or 1
    const int l15  = lane & 15;

    // Cooperative B staging: thread -> column n=tid>>2 (0..63), 16B quarter tid&3
    const int bn = tid >> 2;
    const int bq = tid & 3;
    const uint16_t* gB = Bw + (size_t)(col0 + bn) * ldb + bq * 8;
    uint8_t* lB = &Btile[0][0] + bn * BPAD + bq * 16;
    const size_t bufstride = 64 * BPAD;

    const uint16_t* Abase = A + (size_t)(row0 + l15) * lda + (half << 3);

    auto loadA = [&](Frag16& a, int kk) {
#pragma unroll
        for (int p = 0; p < 8; ++p) {
            // p<4: k = 2p + 8*half ; p>=4: k = 16 + 2(p-4) + 8*half
            const int koff = ((p & 4) << 2) + ((p & 3) << 1);
            a.u[p] = *(const uint32_t*)(Abase + kk + koff);
        }
    };

    const v8f zero8 = {0.f,0.f,0.f,0.f,0.f,0.f,0.f,0.f};
    v8f acc[4];
#pragma unroll
    for (int j = 0; j < 4; ++j) acc[j] = zero8;

    // Prologue: stage k=0 tile, preload first A frag
    Frag16 a_cur;
    loadA(a_cur, 0);
    stage16(gB, lB);
    wait_async0();
    __syncthreads();

    for (int kk = 0; kk < K; kk += 32) {
        const int  cur  = (kk >> 5) & 1;
        const bool more = (kk + 32) < K;

        // Prefetch next tile / next A frag while computing current
        if (more) stage16(gB + (kk + 32), lB + (cur ^ 1) * bufstride);
        Frag16 a_nxt;
        if (more) loadA(a_nxt, kk + 32);

#pragma unroll
        for (int j = 0; j < 4; ++j) {
            // B frag: column col0+16j+l15 -> LDS row (16j+l15), 16B at half*16 and 32+half*16
            const uint32_t* lsrc = (const uint32_t*)
                (&Btile[cur][0] + (16 * j + l15) * BPAD + (half << 4));
            Frag16 b;
#pragma unroll
            for (int p = 0; p < 8; ++p) b.u[p] = lsrc[p + (p & 4)];
            acc[j] = __builtin_amdgcn_wmma_f32_16x16x32_bf16(
                false, a_cur.v, false, b.v, (short)0, acc[j], false, false);
        }

        wait_async0();
        __syncthreads();
        if (more) a_cur = a_nxt;
    }

    // Epilogue: C/D layout — VGPR r, lanes 0-15 => M=r, lanes 16-31 => M=r+8
#pragma unroll
    for (int j = 0; j < 4; ++j) {
        const int n = col0 + 16 * j + l15;
        float badd = bias ? bias[n] : 0.0f;
#pragma unroll
        for (int r = 0; r < 8; ++r) {
            const int m = row0 + r + (half << 3);
            float v = acc[j][r] + badd;
            if (CinitBf) v += bf2f(CinitBf[(size_t)m * ldcin + n]);
            if (Cf)  Cf [(size_t)m * ldc + n] = v;
            if (Cbf) Cbf[(size_t)m * ldc + n] = f2bf(v);
        }
    }
}

// ---------------------------------------------------------------------------
// LSTM gate nonlinearity: z[B,4H] (i,f,g,o), c[B,H] in f32; writes h as bf16
// to one or two destinations (recurrent buffer / enc_out slice).
// ---------------------------------------------------------------------------
__global__ __launch_bounds__(256) void lstm_gates(
    const float* __restrict__ z, float* __restrict__ c,
    uint16_t* __restrict__ h1, int s1,
    uint16_t* __restrict__ h2, long long s2)
{
    const int idx = blockIdx.x * 256 + threadIdx.x;   // over B*H
    if (idx >= BB * HH) return;
    const int b = idx >> 8;        // /H
    const int h = idx & (HH - 1);  // %H
    const float* zb = z + (size_t)b * G4H;
    const float ig = sigmoidf_(zb[h]);
    const float fg = sigmoidf_(zb[HH + h]);
    const float gg = tanhf(zb[2 * HH + h]);
    const float og = sigmoidf_(zb[3 * HH + h]);
    const float cn = fg * c[idx] + ig * gg;
    c[idx] = cn;
    const uint16_t hb = f2bf(og * tanhf(cn));
    h1[(size_t)b * s1 + h] = hb;
    if (h2) h2[(size_t)b * s2 + h] = hb;
}

// ---------------------------------------------------------------------------
// Attention scores: u[b,s] = sum_h tanh(enc_proj[b,s,h] + q[b,h]) * vt[h]
// Block = 8 waves; wave handles one s; lane handles 8 contiguous h.
// ---------------------------------------------------------------------------
__global__ __launch_bounds__(256) void attn_u(
    const uint16_t* __restrict__ encproj, const float* __restrict__ q,
    const float* __restrict__ vt, float* __restrict__ u)
{
    const int b    = blockIdx.y;
    const int s    = blockIdx.x * 8 + (threadIdx.x >> 5);
    const int lane = threadIdx.x & 31;

    float qv[8], vv[8];
    const float* qb = q + (size_t)b * HH + lane * 8;
#pragma unroll
    for (int j = 0; j < 8; ++j) { qv[j] = qb[j]; vv[j] = vt[lane * 8 + j]; }

    const uint16_t* p = encproj + ((size_t)b * SS + s) * HH + lane * 8;
    float acc = 0.0f;
#pragma unroll
    for (int j = 0; j < 8; ++j)
        acc += tanhf(bf2f(p[j]) + qv[j]) * vv[j];

#pragma unroll
    for (int off = 16; off; off >>= 1)
        acc += __shfl_xor(acc, off, 32);
    if (lane == 0) u[(size_t)b * SS + s] = acc;
}

// ---------------------------------------------------------------------------
// Softmax over u[b,:] (S=512, 2 per thread); writes probs in-place and
// into d_out[b, t, :].
// ---------------------------------------------------------------------------
__global__ __launch_bounds__(256) void softmax_write(
    float* __restrict__ u, float* __restrict__ out, int t)
{
    const int b = blockIdx.x;
    const int tid = threadIdx.x;
    const int wave = tid >> 5, lane = tid & 31;
    float* ub = u + (size_t)b * SS;
    float a0 = ub[tid], a1 = ub[tid + 256];

    __shared__ float sm[8];
    __shared__ float ssum[8];

    float m = fmaxf(a0, a1);
#pragma unroll
    for (int off = 16; off; off >>= 1) m = fmaxf(m, __shfl_xor(m, off, 32));
    if (lane == 0) sm[wave] = m;
    __syncthreads();
    float mm = sm[0];
#pragma unroll
    for (int w = 1; w < 8; ++w) mm = fmaxf(mm, sm[w]);

    float e0 = __expf(a0 - mm), e1 = __expf(a1 - mm);
    float s = e0 + e1;
#pragma unroll
    for (int off = 16; off; off >>= 1) s += __shfl_xor(s, off, 32);
    if (lane == 0) ssum[wave] = s;
    __syncthreads();
    float tot = 0.0f;
#pragma unroll
    for (int w = 0; w < 8; ++w) tot += ssum[w];
    const float inv = 1.0f / tot;

    const float p0 = e0 * inv, p1 = e1 * inv;
    ub[tid] = p0; ub[tid + 256] = p1;
    float* ob = out + ((size_t)b * SS + t) * SS;
    ob[tid] = p0; ob[tid + 256] = p1;
}

// ---------------------------------------------------------------------------
// Context: din[b,h] = sum_s probs[b,s] * enc_out[b,s,h]; write bf16 into the
// decoder concat buffer (cols 0..255).
// ---------------------------------------------------------------------------
__global__ __launch_bounds__(256) void context_din(
    const float* __restrict__ probs, const uint16_t* __restrict__ encoutbf,
    uint16_t* __restrict__ catbuf)
{
    const int b = blockIdx.x, h = threadIdx.x;
    const float* pb = probs + (size_t)b * SS;
    const uint16_t* eb = encoutbf + (size_t)b * SS * HH + h;
    float acc = 0.0f;
    for (int s = 0; s < SS; ++s)
        acc += pb[s] * bf2f(eb[(size_t)s * HH]);
    catbuf[(size_t)b * 512 + h] = f2bf(acc);
}

// ---------------------------------------------------------------------------
// Small utility kernels
// ---------------------------------------------------------------------------
__global__ void f32_to_bf16_k(const float* __restrict__ in,
                              uint16_t* __restrict__ out, int n)
{
    const int i = blockIdx.x * 256 + threadIdx.x;
    if (i < n) out[i] = f2bf(in[i]);
}

__global__ void build_wcat_k(const float* __restrict__ wih,
                             const float* __restrict__ whh,
                             uint16_t* __restrict__ wcat, int n /* = 4H*512 */)
{
    const int i = blockIdx.x * 256 + threadIdx.x;
    if (i >= n) return;
    const int row = i >> 9;         // /512
    const int k   = i & 511;        // %512
    const float v = (k < DD) ? wih[(size_t)row * DD + k]
                             : whh[(size_t)row * HH + (k - DD)];
    wcat[i] = f2bf(v);
}

__global__ void fill_f32_k(float* p, int n, float v)
{
    const int i = blockIdx.x * 256 + threadIdx.x;
    if (i < n) p[i] = v;
}
__global__ void fill_u16_k(uint16_t* p, int n, uint16_t v)
{
    const int i = blockIdx.x * 256 + threadIdx.x;
    if (i < n) p[i] = v;
}
__global__ void copy_h_to_cat_k(const uint16_t* __restrict__ hbf,
                                uint16_t* __restrict__ cat)
{
    const int i = blockIdx.x * 256 + threadIdx.x;  // over B*H
    const int b = i >> 8, h = i & 255;
    cat[(size_t)b * 512 + 256 + h] = hbf[i];
}

// ---------------------------------------------------------------------------
extern "C" void kernel_launch(void* const* d_in, const int* in_sizes, int n_in,
                              void* d_out, int out_size, void* d_ws, size_t ws_size,
                              hipStream_t stream)
{
    (void)in_sizes; (void)n_in; (void)out_size; (void)ws_size;
    const float* x       = (const float*)d_in[0];
    const float* enc_Wih = (const float*)d_in[1];
    const float* enc_Whh = (const float*)d_in[2];
    const float* enc_b   = (const float*)d_in[3];
    const float* dec_Wih = (const float*)d_in[4];
    const float* dec_Whh = (const float*)d_in[5];
    const float* dec_b   = (const float*)d_in[6];
    const float* W1_w    = (const float*)d_in[7];
    const float* W1_b    = (const float*)d_in[8];
    const float* W2_w    = (const float*)d_in[9];
    const float* W2_b    = (const float*)d_in[10];
    const float* vt_w    = (const float*)d_in[11];
    float* out = (float*)d_out;

    // ---- carve workspace ----
    char* wp = (char*)d_ws;
    auto carve = [&](size_t bytes) -> void* {
        void* r = (void*)wp;
        wp += (bytes + 255) & ~(size_t)255;
        return r;
    };
    uint16_t* x_bf      = (uint16_t*)carve((size_t)BB * SS * DD * 2);   // 33.5 MB
    uint16_t* XWbf      = (uint16_t*)carve((size_t)BB * SS * G4H * 2);  // 134 MB
    uint16_t* encoutbf  = (uint16_t*)carve((size_t)BB * SS * HH * 2);   // 33.5 MB
    uint16_t* encprojbf = (uint16_t*)carve((size_t)BB * SS * HH * 2);   // 33.5 MB
    uint16_t* wih_bf    = (uint16_t*)carve((size_t)G4H * DD * 2);
    uint16_t* whh_bf    = (uint16_t*)carve((size_t)G4H * HH * 2);
    uint16_t* wcat_bf   = (uint16_t*)carve((size_t)G4H * 512 * 2);
    uint16_t* w1_bf     = (uint16_t*)carve((size_t)HH * HH * 2);
    uint16_t* w2_bf     = (uint16_t*)carve((size_t)HH * HH * 2);
    float*    z         = (float*)carve((size_t)BB * G4H * 4);
    float*    c         = (float*)carve((size_t)BB * HH * 4);
    uint16_t* hbf       = (uint16_t*)carve((size_t)BB * HH * 2);
    uint16_t* catbuf    = (uint16_t*)carve((size_t)BB * 512 * 2);
    float*    q         = (float*)carve((size_t)BB * HH * 4);
    float*    u         = (float*)carve((size_t)BB * SS * 4);

    auto conv = [&](const float* src, uint16_t* dst, int n) {
        f32_to_bf16_k<<<(n + 255) / 256, 256, 0, stream>>>(src, dst, n);
    };
    auto gemm = [&](const uint16_t* A, int lda, const uint16_t* Bw, int ldb,
                    const float* bias, const uint16_t* Cinit, long long ldcin,
                    float* Cf, uint16_t* Cbf, long long ldc,
                    int M, int N, int K) {
        dim3 g(N / 64, (M + 127) / 128);
        gemm_bf16_wmma<<<g, 256, 0, stream>>>(A, lda, Bw, ldb, bias, Cinit,
                                              ldcin, Cf, Cbf, ldc, M, N, K);
    };

    // ---- one-time conversions ----
    conv(x, x_bf, BB * SS * DD);
    conv(enc_Wih, wih_bf, G4H * DD);
    conv(enc_Whh, whh_bf, G4H * HH);
    conv(W1_w, w1_bf, HH * HH);
    conv(W2_w, w2_bf, HH * HH);
    build_wcat_k<<<(G4H * 512 + 255) / 256, 256, 0, stream>>>(dec_Wih, dec_Whh,
                                                              wcat_bf, G4H * 512);

    // ---- big parallel GEMM: XW = x @ enc_Wih^T + enc_b  (bf16 out) ----
    gemm(x_bf, DD, wih_bf, DD, enc_b, nullptr, 0,
         nullptr, XWbf, G4H, BB * SS, G4H, DD);

    // ---- encoder recurrence ----
    fill_f32_k<<<(BB * HH + 255) / 256, 256, 0, stream>>>(c, BB * HH, 0.0f);
    fill_u16_k<<<(BB * HH + 255) / 256, 256, 0, stream>>>(hbf, BB * HH, 0);
    for (int t = 0; t < SS; ++t) {
        // z = h @ Whh^T + XW[:, t, :]
        gemm(hbf, HH, whh_bf, HH, nullptr,
             XWbf + (size_t)t * G4H, (long long)SS * G4H,
             z, nullptr, G4H, BB, G4H, HH);
        lstm_gates<<<(BB * HH + 255) / 256, 256, 0, stream>>>(
            z, c, hbf, HH, encoutbf + (size_t)t * HH, (long long)SS * HH);
    }

    // ---- enc_proj = enc_out @ W1^T + W1_b  (bf16) ----
    gemm(encoutbf, HH, w1_bf, HH, W1_b, nullptr, 0,
         nullptr, encprojbf, HH, BB * SS, HH, HH);

    // ---- decoder init: din = 0, h = encoder final h, c carries over ----
    fill_u16_k<<<(BB * 512 + 255) / 256, 256, 0, stream>>>(catbuf, BB * 512, 0);
    copy_h_to_cat_k<<<(BB * HH + 255) / 256, 256, 0, stream>>>(hbf, catbuf);

    // ---- pointer decode loop ----
    for (int t = 0; t < SS; ++t) {
        // z = [din|h] @ [dec_Wih|dec_Whh]^T + dec_b
        gemm(catbuf, 512, wcat_bf, 512, dec_b, nullptr, 0,
             z, nullptr, G4H, BB, G4H, 512);
        lstm_gates<<<(BB * HH + 255) / 256, 256, 0, stream>>>(
            z, c, catbuf + 256, 512, nullptr, 0);
        // q = h @ W2^T + W2_b   (A = h half of catbuf, lda=512)
        gemm(catbuf + 256, 512, w2_bf, HH, W2_b, nullptr, 0,
             q, nullptr, HH, BB, HH, HH);
        // u[b,s] = sum_h tanh(enc_proj + q) * vt
        attn_u<<<dim3(SS / 8, BB), 256, 0, stream>>>(encprojbf, q, vt_w, u);
        // softmax -> probs (in-place u) + d_out[b,t,:]
        softmax_write<<<BB, 256, 0, stream>>>(u, out, t);
        // din = probs @ enc_out -> catbuf din half (bf16)
        context_din<<<BB, 256, 0, stream>>>(u, encoutbf, catbuf);
    }
}
